// Net_85298050499236
// MI455X (gfx1250) — matrix operand order repaired and checked by reading, not compiled, not run on hardware
//
#include <hip/hip_runtime.h>
#include <stdint.h>

#define N_NODES 16384
#define N_GRAPH 256
#define NPG 64
#define N_EDGE 131072
#define EL (N_EDGE + N_NODES)   // edges including self loops

typedef __attribute__((ext_vector_type(16))) __bf16 v16bf;
typedef __attribute__((ext_vector_type(8)))  float  v8f;

struct BfFrag { float4 lo, hi; };   // 32 bytes == v16bf

static __device__ __forceinline__ unsigned enc_f(float f) {
  unsigned u = __float_as_uint(f);
  return (u >> 31) ? ~u : (u | 0x80000000u);
}
static __device__ __forceinline__ float dec_f(unsigned e) {
  unsigned u = (e >> 31) ? (e & 0x7FFFFFFFu) : ~e;
  return __uint_as_float(u);
}

// ---------------------------------------------------------------------------
// Pack fp32 activations -> bf16 row-major [rows x K]
// ---------------------------------------------------------------------------
__global__ void k_packA(const float* __restrict__ in, int lda,
                        __bf16* __restrict__ out, int rows, int K) {
  int t = blockIdx.x * 256 + threadIdx.x;
  if (t >= rows * K) return;
  int r = t / K, c = t - r * K;
  out[t] = (__bf16)in[(size_t)r * lda + c];
}

// ---------------------------------------------------------------------------
// Pack fp32 weights (K x M row-major) into B-fragment order:
//   out[((k>>5)*(M>>4) + (m>>4))*512 + lane*16 + e],
//   lane = (m&15) | ((((k&31)>>3)&1)<<4),  e = (k&7) | (((k&31)>>4)<<3)
// GEMM then reads a lane's whole fragment as 2 contiguous b128 loads.
// ---------------------------------------------------------------------------
__global__ void k_packW(const float* __restrict__ W,
                        __bf16* __restrict__ out, int K, int M) {
  int t = blockIdx.x * 256 + threadIdx.x;
  if (t >= K * M) return;
  int k = t / M, m = t - k * M;
  int kblk = k >> 5, krel = k & 31;
  int lane = (m & 15) | (((krel >> 3) & 1) << 4);
  int e = (krel & 7) | ((krel >> 4) << 3);
  out[((size_t)kblk * (M >> 4) + (m >> 4)) * 512 + lane * 16 + e] = (__bf16)W[t];
}

// ---------------------------------------------------------------------------
// WMMA GEMM on packed bf16: C[rows x (16*TN per wave)] = act(A @ W + bias [+C])
// rows%16==0, K%32==0, M%(16*TN)==0.  FLAGS: bit0 = relu, bit1 = accumulate
// ---------------------------------------------------------------------------
template <int TN, int FLAGS, bool HB>
__global__ __launch_bounds__(256)
void k_gemm(const __bf16* __restrict__ A,     // rows x K, bf16 row-major
            const __bf16* __restrict__ Wp,    // packed fragments
            const float* __restrict__ bias,
            float* __restrict__ C, int ldc,
            int rows, int K, int M)
{
  const int wave = threadIdx.x >> 5;
  const int lane = threadIdx.x & 31;
  const int mt = M / (16 * TN);
  int tile = blockIdx.x * 8 + wave;
  if (tile >= (rows >> 4) * mt) return;
  const int tm = tile / mt, tn = tile - tm * mt;
  const int row0 = tm << 4, col0 = tn * (16 * TN);

  const int ar_ = lane & 15;            // A row within tile
  const int akb = (lane >> 4) << 3;     // A K-base: 0 or 8
  const __bf16* arow = A + (size_t)(row0 + ar_) * K + akb;
  const int mtiles = M >> 4;

  v8f acc[TN];
  const v8f vzero = {0.f, 0.f, 0.f, 0.f, 0.f, 0.f, 0.f, 0.f};
#pragma unroll
  for (int t = 0; t < TN; ++t) acc[t] = vzero;

  for (int k0 = 0; k0 < K; k0 += 32) {
    BfFrag af;
    af.lo = *(const float4*)(arow + k0);        // K = kb .. kb+7
    af.hi = *(const float4*)(arow + k0 + 16);   // K = kb+16 .. kb+23
    v16bf av = __builtin_bit_cast(v16bf, af);

    const __bf16* bp = Wp + ((size_t)(k0 >> 5) * mtiles + (col0 >> 4)) * 512 + lane * 16;
#pragma unroll
    for (int t = 0; t < TN; ++t) {
      BfFrag bf_;
      bf_.lo = *(const float4*)(bp + (size_t)t * 512);
      bf_.hi = *(const float4*)(bp + (size_t)t * 512 + 8);
      v16bf bv = __builtin_bit_cast(v16bf, bf_);
      acc[t] = __builtin_amdgcn_wmma_f32_16x16x32_bf16(
          false, av, false, bv, (short)0, acc[t], false, false);
    }
  }

  const int rb = row0 + ((lane >> 4) << 3);
#pragma unroll
  for (int t = 0; t < TN; ++t) {
    const int ccol = col0 + t * 16 + (lane & 15);
    const float bb = HB ? bias[ccol] : 0.0f;
#pragma unroll
    for (int j = 0; j < 8; ++j) {
      float v = acc[t][j] + bb;
      float* cp = C + (size_t)(rb + j) * ldc + ccol;
      if (FLAGS & 2) v += *cp;
      if (FLAGS & 1) v = fmaxf(v, 0.0f);
      *cp = v;
    }
  }
}

// ---------------------------------------------------------------------------
// Elementwise / fill / copy
// ---------------------------------------------------------------------------
__global__ void k_fill_f32(float* p, float v, int n) {
  int i = blockIdx.x * 256 + threadIdx.x; if (i < n) p[i] = v;
}
__global__ void k_fill_u32(unsigned* p, unsigned v, int n) {
  int i = blockIdx.x * 256 + threadIdx.x; if (i < n) p[i] = v;
}
__global__ void k_copy(float* d, const float* s, int n) {
  int i = blockIdx.x * 256 + threadIdx.x; if (i < n) d[i] = s[i];
}
__global__ void k_bias_relu(float* v, const float* b, int M, int total) {
  int t = blockIdx.x * 256 + threadIdx.x;
  if (t < total) { int c = t % M; v[t] = fmaxf(v[t] + b[c], 0.0f); }
}

// ---------------------------------------------------------------------------
// BatchNorm (training-mode batch stats over N rows, per column)
// ---------------------------------------------------------------------------
__global__ void k_bn_stats(const float* __restrict__ v, int M,
                           const float* g, const float* b,
                           float* scale, float* shift)
{
  int c = blockIdx.x;
  __shared__ float ss[256], ss2[256];
  float s = 0.f, s2 = 0.f;
  for (int i = threadIdx.x; i < N_NODES; i += 256) {
    float x = v[(size_t)i * M + c]; s += x; s2 += x * x;
  }
  ss[threadIdx.x] = s; ss2[threadIdx.x] = s2; __syncthreads();
  for (int st = 128; st > 0; st >>= 1) {
    if (threadIdx.x < st) { ss[threadIdx.x] += ss[threadIdx.x+st]; ss2[threadIdx.x] += ss2[threadIdx.x+st]; }
    __syncthreads();
  }
  if (threadIdx.x == 0) {
    float m = ss[0] / N_NODES;
    float var = ss2[0] / N_NODES - m * m;
    float sc = g[c] * rsqrtf(var + 1e-5f);
    scale[c] = sc; shift[c] = b[c] - m * sc;
  }
}
__global__ void k_bn_apply(float* v, const float* scale, const float* shift, int M, int total) {
  int t = blockIdx.x * 256 + threadIdx.x;
  if (t < total) { int c = t % M; v[t] = v[t] * scale[c] + shift[c]; }
}

// ---------------------------------------------------------------------------
// Mean pool per graph (nodes contiguous: graph g = rows [g*64, g*64+64))
// ---------------------------------------------------------------------------
__global__ void k_pool(const float* __restrict__ v, int ld, int width,
                       float* __restrict__ P, int off)
{
  int t = blockIdx.x * 256 + threadIdx.x;
  if (t >= N_GRAPH * width) return;
  int g = t / width, c = t - g * width;
  const float* p = v + (size_t)g * NPG * ld + c;
  float s = 0.f;
  for (int i = 0; i < NPG; ++i) s += p[(size_t)i * ld];
  P[(size_t)g * 608 + off + c] = s * (1.0f / NPG);
}

// ---------------------------------------------------------------------------
// GCN: degree / norm / scatter
// ---------------------------------------------------------------------------
__global__ void k_deg_count(const long long* __restrict__ ei, float* deg) {
  int e = blockIdx.x * 256 + threadIdx.x;
  if (e < N_EDGE) atomicAdd(&deg[(int)ei[N_EDGE + e]], 1.0f);
}
__global__ void k_dinv(float* deg) {
  int n = blockIdx.x * 256 + threadIdx.x;
  if (n < N_NODES) deg[n] = rsqrtf(fmaxf(deg[n], 1.0f));
}
__global__ void k_gcn_self(const float* __restrict__ H, const float* __restrict__ dinv,
                           float* __restrict__ out, int M) {
  int t = blockIdx.x * 256 + threadIdx.x;
  if (t < N_NODES * M) { int n = t / M; float w = dinv[n]; out[t] = H[t] * w * w; }
}
__global__ void k_gcn_scatter(const long long* __restrict__ ei, const float* __restrict__ H,
                              const float* __restrict__ dinv, float* __restrict__ out, int M) {
  int t = blockIdx.x * 256 + threadIdx.x;
  if (t >= N_EDGE * M) return;
  int e = t / M, c = t - e * M;
  int s = (int)ei[e], d = (int)ei[N_EDGE + e];
  atomicAdd(&out[(size_t)d * M + c], H[(size_t)s * M + c] * dinv[s] * dinv[d]);
}

// ---------------------------------------------------------------------------
// GAT
// ---------------------------------------------------------------------------
__global__ void k_gat_alr(const float* __restrict__ H, const float* __restrict__ as_,
                          const float* __restrict__ ad_, float* al, float* ar, int C) {
  int t = blockIdx.x * 256 + threadIdx.x;
  if (t >= N_NODES * 3) return;
  int n = t / 3, h = t - n * 3;
  const float* hp = H + (size_t)n * 3 * C + h * C;
  const float* ap = as_ + h * C;
  const float* dp = ad_ + h * C;
  float sa = 0.f, sd = 0.f;
  for (int c = 0; c < C; ++c) { float x = hp[c]; sa += x * ap[c]; sd += x * dp[c]; }
  al[t] = sa; ar[t] = sd;
}
__global__ void k_gat_e(const long long* __restrict__ ei, const float* al, const float* ar,
                        float* ebuf, unsigned* mx) {
  int t = blockIdx.x * 256 + threadIdx.x;
  if (t >= EL) return;
  int s, d;
  if (t < N_EDGE) { s = (int)ei[t]; d = (int)ei[N_EDGE + t]; } else { s = d = t - N_EDGE; }
  for (int h = 0; h < 3; ++h) {
    float e = al[s * 3 + h] + ar[d * 3 + h];
    e = (e > 0.f) ? e : 0.2f * e;                // leaky_relu(0.2)
    ebuf[t * 3 + h] = e;
    atomicMax(&mx[d * 3 + h], enc_f(e));
  }
}
__global__ void k_gat_exden(const long long* __restrict__ ei, float* ebuf,
                            const unsigned* mx, float* den) {
  int t = blockIdx.x * 256 + threadIdx.x;
  if (t >= EL) return;
  int d;
  if (t < N_EDGE) { d = (int)ei[N_EDGE + t]; } else { d = t - N_EDGE; }
  for (int h = 0; h < 3; ++h) {
    float ex = __expf(ebuf[t * 3 + h] - dec_f(mx[d * 3 + h]));
    ebuf[t * 3 + h] = ex;
    atomicAdd(&den[d * 3 + h], ex);
  }
}
__global__ void k_gat_scatter(const long long* __restrict__ ei, const float* __restrict__ H,
                              const float* __restrict__ ebuf, const float* __restrict__ den,
                              float* __restrict__ out, int C) {
  int t = blockIdx.x * 256 + threadIdx.x;
  int HC = 3 * C;
  if (t >= EL * HC) return;
  int idx = t / HC, rem = t - idx * HC;
  int h = rem / C;
  int s, d;
  if (idx < N_EDGE) { s = (int)ei[idx]; d = (int)ei[N_EDGE + idx]; } else { s = d = idx - N_EDGE; }
  float alpha = ebuf[idx * 3 + h] / (den[d * 3 + h] + 1e-16f);
  atomicAdd(&out[(size_t)d * HC + rem], H[(size_t)s * HC + rem] * alpha);
}

// ---------------------------------------------------------------------------
// GIN scatter (no self loops)
// ---------------------------------------------------------------------------
__global__ void k_gin_scatter(const long long* __restrict__ ei, const float* __restrict__ in,
                              float* __restrict__ out, int M) {
  int t = blockIdx.x * 256 + threadIdx.x;
  if (t >= N_EDGE * M) return;
  int e = t / M, c = t - e * M;
  int s = (int)ei[e], d = (int)ei[N_EDGE + e];
  atomicAdd(&out[(size_t)d * M + c], in[(size_t)s * M + c]);
}

// ---------------------------------------------------------------------------
// Graphormer
// ---------------------------------------------------------------------------
__global__ void k_deg_int(const long long* __restrict__ ei, int* indeg, int* outdeg) {
  int e = blockIdx.x * 256 + threadIdx.x;
  if (e < N_EDGE) {
    atomicAdd(&outdeg[(int)ei[e]], 1);
    atomicAdd(&indeg[(int)ei[N_EDGE + e]], 1);
  }
}
__global__ void k_add_emb(float* h, const int* indeg, const int* outdeg,
                          const float* ie, const float* oe) {
  int t = blockIdx.x * 256 + threadIdx.x;
  if (t >= N_NODES * 256) return;
  int n = t >> 8, c = t & 255;
  h[t] += ie[min(indeg[n], 3) * 256 + c] + oe[min(outdeg[n], 3) * 256 + c];
}
__global__ void k_ln(const float* __restrict__ in, float* __restrict__ out,
                     const float* g, const float* b, int d) {
  int row = blockIdx.x * 256 + threadIdx.x;
  if (row >= N_NODES) return;
  const float* p = in + (size_t)row * d;
  float s = 0.f, s2 = 0.f;
  for (int c = 0; c < d; ++c) { float v = p[c]; s += v; s2 += v * v; }
  float m = s / d, var = s2 / d - m * m;
  float r = rsqrtf(var + 1e-5f);
  float* o = out + (size_t)row * d;
  for (int c = 0; c < d; ++c) o[c] = (p[c] - m) * r * g[c] + b[c];
}
// one block = (graph b, head h); 64 threads = 64 queries
__global__ __launch_bounds__(64)
void k_attn(const float* __restrict__ qkv,   // [N,768] = q|k|v
            const int* __restrict__ sp,      // [B,64,64]
            const float* __restrict__ tab,   // [5,16]
            float* __restrict__ out)         // [N,256]
{
  int b = blockIdx.x >> 4;
  int h = blockIdx.x & 15;
  int t = threadIdx.x;
  __shared__ float Ks[NPG * 16], Vs[NPG * 16];
  const float* base = qkv + (size_t)b * NPG * 768;
  for (int d = 0; d < 16; ++d) {
    Ks[t * 16 + d] = base[(size_t)t * 768 + 256 + h * 16 + d];
    Vs[t * 16 + d] = base[(size_t)t * 768 + 512 + h * 16 + d];
  }
  __syncthreads();
  float q[16];
  for (int d = 0; d < 16; ++d) q[d] = base[(size_t)t * 768 + h * 16 + d];
  const int* spr = sp + ((size_t)b * NPG + t) * NPG;
  float sc[NPG];
  float mxv = -1e30f;
  for (int kk = 0; kk < NPG; ++kk) {
    float s = 0.f;
    for (int d = 0; d < 16; ++d) s += q[d] * Ks[kk * 16 + d];
    s = s * 0.25f + tab[spr[kk] * 16 + h];     // 1/sqrt(16) = 0.25
    sc[kk] = s; mxv = fmaxf(mxv, s);
  }
  float den = 0.f;
  for (int kk = 0; kk < NPG; ++kk) { sc[kk] = __expf(sc[kk] - mxv); den += sc[kk]; }
  float inv = 1.0f / den;
  float* op = out + ((size_t)b * NPG + t) * 256 + h * 16;
  for (int d = 0; d < 16; ++d) {
    float a = 0.f;
    for (int kk = 0; kk < NPG; ++kk) a += sc[kk] * Vs[kk * 16 + d];
    op[d] = a * inv;
  }
}

// Final FC3: 64 -> 1, relu
__global__ void k_fc3(const float* __restrict__ F2, const float* __restrict__ W,
                      const float* __restrict__ b, float* __restrict__ out) {
  int t = blockIdx.x * 256 + threadIdx.x;
  if (t >= N_GRAPH) return;
  float s = 0.f;
  for (int c = 0; c < 64; ++c) s += F2[(size_t)t * 64 + c] * W[c];
  out[t] = fmaxf(s + b[0], 0.0f);
}

// ---------------------------------------------------------------------------
// Host orchestration
// ---------------------------------------------------------------------------
extern "C" void kernel_launch(void* const* d_in, const int* in_sizes, int n_in,
                              void* d_out, int out_size, void* d_ws, size_t ws_size,
                              hipStream_t stream)
{
  (void)in_sizes; (void)n_in; (void)out_size; (void)ws_size;
  auto f  = [&](int i) { return (const float*)d_in[i]; };
  const float*     x  = f(0);
  const long long* ei = (const long long*)d_in[1];
  const int*       sp = (const int*)d_in[3];

  char* ws = (char*)d_ws;
  float*    A    = (float*)(ws);                 // N x 768
  float*    Bb   = (float*)(ws + 50331648);      // N x 768
  float*    Cc   = (float*)(ws + 100663296);     // N x 384
  float*    ebuf = (float*)(ws + 125829120);     // (E+N) x 3
  unsigned* mx   = (unsigned*)(ws + 127598592);  // N x 3
  float*    den  = (float*)(ws + 127795200);     // N x 3
  float*    al   = (float*)(ws + 127991808);     // N x 3
  float*    ar   = (float*)(ws + 128188416);     // N x 3
  float*    dinv = (float*)(ws + 128385024);     // N
  int*      indeg  = (int*)(ws + 128450560);     // N
  int*      outdeg = (int*)(ws + 128516096);     // N
  float*    bnsc = (float*)(ws + 128581632);     // 768
  float*    bnsh = (float*)(ws + 128584704);     // 768
  float*    P    = (float*)(ws + 128587776);     // 256 x 608
  float*    F1   = (float*)(ws + 129210368);     // 256 x 128
  float*    F2   = (float*)(ws + 129341440);     // 256 x 64
  __bf16*   Apk  = (__bf16*)(ws + 129406976);    // N x 768 bf16  (25165824 B)
  __bf16*   Wpk  = (__bf16*)(ws + 154572800);    // up to 768*768 bf16

  auto EW = [](int n) { return (n + 255) / 256; };
  // GEMM wrapper: packs A (optional) + W, then launches WMMA kernel.
  auto gemm = [&](const float* Ap, int lda, const float* Wp, const float* bp,
                  float* Cp, int ldc, int rows, int K, int M, int flags,
                  bool repackA = true) {
    if (repackA)
      k_packA<<<EW(rows * K), 256, 0, stream>>>(Ap, lda, Apk, rows, K);
    k_packW<<<EW(K * M), 256, 0, stream>>>(Wp, Wpk, K, M);
    const int tnw = (M % 64 == 0) ? 4 : 2;
    const int tiles = (rows / 16) * (M / (16 * tnw));
    const int blocks = (tiles + 7) / 8;
#define GLAUNCH(TN, FL, HB) \
    k_gemm<TN, FL, HB><<<blocks, 256, 0, stream>>>(Apk, Wpk, bp, Cp, ldc, rows, K, M)
    if (tnw == 4) {
      if (bp) {
        if (flags == 0) GLAUNCH(4, 0, true);
        else if (flags == 1) GLAUNCH(4, 1, true);
        else GLAUNCH(4, 2, true);
      } else {
        if (flags == 0) GLAUNCH(4, 0, false);
        else if (flags == 1) GLAUNCH(4, 1, false);
        else GLAUNCH(4, 2, false);
      }
    } else {
      if (bp) {
        if (flags == 0) GLAUNCH(2, 0, true);
        else if (flags == 1) GLAUNCH(2, 1, true);
        else GLAUNCH(2, 2, true);
      } else {
        if (flags == 0) GLAUNCH(2, 0, false);
        else if (flags == 1) GLAUNCH(2, 1, false);
        else GLAUNCH(2, 2, false);
      }
    }
#undef GLAUNCH
  };
  auto bn = [&](float* v, int M, int gi, int bi) {
    k_bn_stats<<<M, 256, 0, stream>>>(v, M, f(gi), f(bi), bnsc, bnsh);
    k_bn_apply<<<EW(N_NODES * M), 256, 0, stream>>>(v, bnsc, bnsh, M, N_NODES * M);
  };

  // ===== GCN degree norm (with self loops: init deg=1) =====
  k_fill_f32<<<EW(N_NODES), 256, 0, stream>>>(dinv, 1.0f, N_NODES);
  k_deg_count<<<EW(N_EDGE), 256, 0, stream>>>(ei, dinv);
  k_dinv<<<EW(N_NODES), 256, 0, stream>>>(dinv);

  auto gcn_layer = [&](const float* in, int lda, int K, int M,
                       int wi, int bi, int bng, int bnb, float* H, float* out) {
    gemm(in, lda, f(wi), nullptr, H, M, N_NODES, K, M, 0);
    k_gcn_self<<<EW(N_NODES * M), 256, 0, stream>>>(H, dinv, out, M);
    k_gcn_scatter<<<EW(N_EDGE * M), 256, 0, stream>>>(ei, H, dinv, out, M);
    k_bias_relu<<<EW(N_NODES * M), 256, 0, stream>>>(out, f(bi), M, N_NODES * M);
    bn(out, M, bng, bnb);
  };
  gcn_layer(x,  64,  64, 256, 4, 5, 10, 11, A, Bb);
  gcn_layer(Bb, 256, 256, 128, 6, 7, 12, 13, A, Cc);
  gcn_layer(Cc, 128, 128, 128, 8, 9, 14, 15, A, Bb);
  k_pool<<<EW(N_GRAPH * 128), 256, 0, stream>>>(Bb, 128, 128, P, 0);

  // ===== GAT branch =====
  auto gat_layer = [&](const float* in, int lda, int K, int C,
                       int wi, int asi, int adi, int bi, int bng, int bnb,
                       float* H, float* out) {
    int M = 3 * C;
    gemm(in, lda, f(wi), nullptr, H, M, N_NODES, K, M, 0);
    k_gat_alr<<<EW(N_NODES * 3), 256, 0, stream>>>(H, f(asi), f(adi), al, ar, C);
    k_fill_u32<<<EW(N_NODES * 3), 256, 0, stream>>>(mx, 0u, N_NODES * 3);
    k_gat_e<<<EW(EL), 256, 0, stream>>>(ei, al, ar, ebuf, mx);
    k_fill_f32<<<EW(N_NODES * 3), 256, 0, stream>>>(den, 0.0f, N_NODES * 3);
    k_gat_exden<<<EW(EL), 256, 0, stream>>>(ei, ebuf, mx, den);
    k_fill_f32<<<EW(N_NODES * M), 256, 0, stream>>>(out, 0.0f, N_NODES * M);
    k_gat_scatter<<<EW(EL * M), 256, 0, stream>>>(ei, H, ebuf, den, out, C);
    k_bias_relu<<<EW(N_NODES * M), 256, 0, stream>>>(out, f(bi), M, N_NODES * M);
    bn(out, M, bng, bnb);
  };
  gat_layer(x,  64,  64, 256, 28, 29, 30, 31, 16, 17, A, Bb);
  gat_layer(Bb, 768, 768, 128, 32, 33, 34, 35, 18, 19, A, Cc);
  gat_layer(Cc, 384, 384, 128, 36, 37, 38, 39, 20, 21, A, Bb);
  k_pool<<<EW(N_GRAPH * 384), 256, 0, stream>>>(Bb, 384, 384, P, 128);

  // ===== GIN branch (no self loops) =====
  auto gin_layer = [&](const float* in, int K, int H1,
                       int w1, int b1, int w2, int b2, int bng, int bnb,
                       float* agg, float* mid, float* out) {
    k_copy<<<EW(N_NODES * K), 256, 0, stream>>>(agg, in, N_NODES * K);
    k_gin_scatter<<<EW(N_EDGE * K), 256, 0, stream>>>(ei, in, agg, K);
    gemm(agg, K, f(w1), f(b1), mid, H1, N_NODES, K, H1, 1);   // relu
    gemm(mid, H1, f(w2), f(b2), out, H1, N_NODES, H1, H1, 1); // branch relu
    bn(out, H1, bng, bnb);
  };
  gin_layer(x,  64, 256, 40, 41, 42, 43, 22, 23, A, Bb, Cc);
  gin_layer(Cc, 256, 128, 44, 45, 46, 47, 24, 25, A, Bb, Cc);
  gin_layer(Cc, 128,  64, 48, 49, 50, 51, 26, 27, A, Bb, Cc);
  k_pool<<<EW(N_GRAPH * 64), 256, 0, stream>>>(Cc, 64, 64, P, 512);

  // ===== Graphormer branch =====
  k_fill_u32<<<EW(N_NODES), 256, 0, stream>>>((unsigned*)indeg, 0u, N_NODES);
  k_fill_u32<<<EW(N_NODES), 256, 0, stream>>>((unsigned*)outdeg, 0u, N_NODES);
  k_deg_int<<<EW(N_EDGE), 256, 0, stream>>>(ei, indeg, outdeg);
  gemm(x, 64, f(52), f(53), A, 256, N_NODES, 64, 256, 0);
  k_add_emb<<<EW(N_NODES * 256), 256, 0, stream>>>(A, indeg, outdeg, f(54), f(55));

  auto gh_layer = [&](int p0) {
    // p0+0..7: Wq,Wqb,Wk,Wkb,Wv,Wvb,Wo,Wob; +8 spatial; +9..12 ln1g,ln1b,ln2g,ln2b;
    // +13..16: ffW1,ffb1,ffW2,ffb2
    k_ln<<<EW(N_NODES), 256, 0, stream>>>(A, Cc, f(p0 + 9), f(p0 + 10), 256);
    gemm(Cc, 256, f(p0 + 0), f(p0 + 1), Bb + 0,   768, N_NODES, 256, 256, 0);        // Q
    gemm(Cc, 256, f(p0 + 2), f(p0 + 3), Bb + 256, 768, N_NODES, 256, 256, 0, false); // K (A reused)
    gemm(Cc, 256, f(p0 + 4), f(p0 + 5), Bb + 512, 768, N_NODES, 256, 256, 0, false); // V (A reused)
    k_attn<<<N_GRAPH * 16, 64, 0, stream>>>(Bb, sp, f(p0 + 8), Cc);
    gemm(Cc, 256, f(p0 + 6), f(p0 + 7), A, 256, N_NODES, 256, 256, 2);               // h += o@Wo+b
    k_ln<<<EW(N_NODES), 256, 0, stream>>>(A, Cc, f(p0 + 11), f(p0 + 12), 256);
    gemm(Cc, 256, f(p0 + 13), f(p0 + 14), Bb, 256, N_NODES, 256, 256, 1);            // relu ff1
    gemm(Bb, 256, f(p0 + 15), f(p0 + 16), A, 256, N_NODES, 256, 256, 2);             // h += ff2
  };
  gh_layer(56);
  gh_layer(73);
  k_ln<<<EW(N_NODES), 256, 0, stream>>>(A, Cc, f(90), f(91), 256);
  gemm(Cc, 256, f(92), f(93), Bb, 32, N_NODES, 256, 32, 0);
  k_pool<<<EW(N_GRAPH * 32), 256, 0, stream>>>(Bb, 32, 32, P, 576);

  // ===== Final FC head =====
  gemm(P, 608, f(94), f(95), F1, 128, N_GRAPH, 608, 128, 1);
  gemm(F1, 128, f(96), f(97), F2, 64, N_GRAPH, 128, 64, 1);
  k_fc3<<<1, 256, 0, stream>>>(F2, f(98), f(99), (float*)d_out);
}
